// AM2_1271310320197
// MI455X (gfx1250) — matrix operand order repaired and checked by reading, not compiled, not run on hardware
//
#include <hip/hip_runtime.h>
#include <math.h>

// ---------------------------------------------------------------------------
// Problem constants (match reference)
// ---------------------------------------------------------------------------
#define BB    512
#define NPTS  200
#define NA    199          // NPTS - 1
#define DD    128
#define NH    8
#define DKH   16
#define HIDD  512
#define NLAY  3
#define BNROW (BB * NPTS)  // 102400
#define CHM   25600        // FF M-chunk (BNROW / 4)
#define NEGV  (-1.0e9f)

typedef _Float16 v16h __attribute__((ext_vector_type(16)));
typedef _Float16 v8h  __attribute__((ext_vector_type(8)));
typedef float    v8f  __attribute__((ext_vector_type(8)));

// ---------------------------------------------------------------------------
// Weight transpose + f32->f16: W[K,N] -> Wt[N,K] (f16)
// ---------------------------------------------------------------------------
__global__ __launch_bounds__(256) void transpose_f16_kernel(
    const float* __restrict__ W, _Float16* __restrict__ Wt, int K, int N)
{
    long i = (long)blockIdx.x * 256 + threadIdx.x;
    if (i >= (long)K * N) return;
    int k = (int)(i / N), n = (int)(i % N);
    Wt[(long)n * K + k] = (_Float16)W[i];
}

// ---------------------------------------------------------------------------
// Embedding: h = x @ emb_W + emb_b   (K = 2, pure VALU); writes f32 + f16
// ---------------------------------------------------------------------------
__global__ __launch_bounds__(256) void embed_kernel(
    const float* __restrict__ x, const float* __restrict__ W,
    const float* __restrict__ bvec, float* __restrict__ Hout,
    _Float16* __restrict__ Hout16, int rows)
{
    long i = (long)blockIdx.x * 256 + threadIdx.x;
    if (i >= (long)rows * DD) return;
    int r = (int)(i >> 7);
    int c = (int)(i & 127);
    float v = x[r * 2] * W[c] + x[r * 2 + 1] * W[DD + c] + bvec[c];
    Hout[i]   = v;
    Hout16[i] = (_Float16)v;
}

// ---------------------------------------------------------------------------
// WMMA GEMM, f16 operands: C[M,N] = act(A[M,K] @ Bt[N,K]^T + bias).
// One wave -> a 16x64 output strip (4 accumulator tiles): the A chunk is
// loaded once per K-step and reused by 4 WMMAs (4 WMMA : 10 b128 loads).
// All four B-tile loads are hoisted ahead of the WMMAs so each tile keeps
// its own registers and the compiler can use partial s_wait_loadcnt,
// overlapping the load clause with the matrix pipe.
// A VGPR layout (ISA 7.12.2, 16-bit A 16x32): lane half h needs contiguous
// halves k = kc+8h..kc+8h+7 and kc+16+8h..+7 -> two b128 loads.
// B (32x16): lane half h needs k = kc+8h..kc+8h+7 and kc+16+8h..+7 at fixed
// column n, contiguous in the transposed N x K weight -> two b128 loads per tile.
// remapA==1: logical A row m -> physical row (m/199)*200 + m%199 + 1.
// C (f32) and C16 (f16) outputs are each optional (nullable).
// ---------------------------------------------------------------------------
__global__ __launch_bounds__(64) void gemm_wmma_kernel(
    const _Float16* __restrict__ A, const _Float16* __restrict__ Bt,
    const float* __restrict__ bias, float* __restrict__ C,
    _Float16* __restrict__ C16,
    int M, int N, int K, int relu, int remapA)
{
    const int wave = threadIdx.x >> 5;            // 0..1
    const int lane = threadIdx.x & 31;
    const int m0   = blockIdx.x * 16;
    const int n0   = blockIdx.y * 128 + wave * 64; // 64 columns per wave
    const int half = lane >> 4;                    // 0: lanes 0-15, 1: 16-31
    const int l15  = lane & 15;

    int am = m0 + l15;
    long arow;
    if (remapA) {
        int bidx = am / NA;
        int nidx = am - bidx * NA;
        arow = (long)(bidx * NPTS + nidx + 1) * K;
    } else {
        arow = (long)am * K;
    }
    long brow[4];
#pragma unroll
    for (int j = 0; j < 4; ++j) brow[j] = (long)(n0 + j * 16 + l15) * K;

    v8f acc[4] = {{}, {}, {}, {}};
    for (int kc = 0; kc < K; kc += 32) {
        // Issue the whole load clause first: 2 A loads + 4x2 B loads.
        v8h a0 = *(const v8h*)(A + arow + kc + (half << 3));
        v8h a1 = *(const v8h*)(A + arow + kc + 16 + (half << 3));
        v8h b0[4], b1[4];
#pragma unroll
        for (int j = 0; j < 4; ++j) {
            b0[j] = *(const v8h*)(Bt + brow[j] + kc + (half << 3));
            b1[j] = *(const v8h*)(Bt + brow[j] + kc + 16 + (half << 3));
        }
        v16h av;
#pragma unroll
        for (int i = 0; i < 8; ++i) { av[i] = a0[i]; av[8 + i] = a1[i]; }
#pragma unroll
        for (int j = 0; j < 4; ++j) {
            v16h bv;
#pragma unroll
            for (int i = 0; i < 8; ++i) { bv[i] = b0[j][i]; bv[8 + i] = b1[j][i]; }
            acc[j] = __builtin_amdgcn_wmma_f32_16x16x32_f16(
                /*neg_a=*/false, av, /*neg_b=*/false, bv,
                /*c_mod=*/(short)0, acc[j], /*reuse_a=*/false, /*reuse_b=*/false);
        }
    }

    // C/D layout per tile: lane column n = tile_n0 + l15 ; VGPR r -> row m0+r+8*half
#pragma unroll
    for (int j = 0; j < 4; ++j) {
        const int   cn   = n0 + j * 16 + l15;
        const float bval = bias ? bias[cn] : 0.0f;
#pragma unroll
        for (int r = 0; r < 8; ++r) {
            int   cm  = m0 + r + (half << 3);
            float val = acc[j][r] + bval;
            if (relu) val = fmaxf(val, 0.0f);
            long  idx = (long)cm * N + cn;
            if (C)   C[idx]   = val;
            if (C16) C16[idx] = (_Float16)val;
        }
    }
}

// ---------------------------------------------------------------------------
// Encoder attention: one block per (batch, head). f16 in, f16 out,
// fp32 math with exact online softmax; K/V head slice staged in LDS.
// ---------------------------------------------------------------------------
__global__ __launch_bounds__(256) void attn_kernel(
    const _Float16* __restrict__ Q, const _Float16* __restrict__ K,
    const _Float16* __restrict__ V, _Float16* __restrict__ O)
{
    __shared__ float sK[NPTS * DKH];
    __shared__ float sV[NPTS * DKH];
    const int  b    = blockIdx.x / NH;
    const int  h    = blockIdx.x % NH;
    const long base = (long)b * NPTS;
    const int  c0   = h * DKH;

    for (int idx = threadIdx.x; idx < NPTS * DKH; idx += 256) {
        int r = idx / DKH, c = idx % DKH;
        sK[idx] = (float)K[(base + r) * DD + c0 + c];
        sV[idx] = (float)V[(base + r) * DD + c0 + c];
    }
    __syncthreads();

    const float scale = 0.25f; // 1/sqrt(16)
    for (int i = threadIdx.x; i < NPTS; i += 256) {
        float qv[DKH];
#pragma unroll
        for (int c = 0; c < DKH; ++c) qv[c] = (float)Q[(base + i) * DD + c0 + c];
        float m = -3.4e38f, ssum = 0.0f, o[DKH];
#pragma unroll
        for (int c = 0; c < DKH; ++c) o[c] = 0.0f;
        for (int j = 0; j < NPTS; ++j) {
            float s = 0.0f;
#pragma unroll
            for (int c = 0; c < DKH; ++c) s += qv[c] * sK[j * DKH + c];
            s *= scale;
            if (s > m) {
                float sc = __expf(m - s);
                ssum = ssum * sc + 1.0f;
#pragma unroll
                for (int c = 0; c < DKH; ++c) o[c] = o[c] * sc + sV[j * DKH + c];
                m = s;
            } else {
                float w = __expf(s - m);
                ssum += w;
#pragma unroll
                for (int c = 0; c < DKH; ++c) o[c] += w * sV[j * DKH + c];
            }
        }
        float inv = 1.0f / ssum;
#pragma unroll
        for (int c = 0; c < DKH; ++c)
            O[(base + i) * DD + c0 + c] = (_Float16)(o[c] * inv);
    }
}

// ---------------------------------------------------------------------------
// Batch-norm over axes (0,1): reduce then apply, x = A + Bres (residual add).
// Apply writes fp32 h and its f16 mirror (next GEMM's A operand).
// ---------------------------------------------------------------------------
__global__ __launch_bounds__(256) void bn_reduce_kernel(
    const float* __restrict__ A, const float* __restrict__ Bres,
    float* __restrict__ stats, int rows)
{
    __shared__ float s1[256], s2[256];
    const int c = blockIdx.x; // channel
    float sum = 0.0f, sq = 0.0f;
    for (int r = threadIdx.x; r < rows; r += 256) {
        float x = A[(long)r * DD + c] + Bres[(long)r * DD + c];
        sum += x; sq += x * x;
    }
    s1[threadIdx.x] = sum; s2[threadIdx.x] = sq;
    __syncthreads();
    for (int st = 128; st > 0; st >>= 1) {
        if (threadIdx.x < st) {
            s1[threadIdx.x] += s1[threadIdx.x + st];
            s2[threadIdx.x] += s2[threadIdx.x + st];
        }
        __syncthreads();
    }
    if (threadIdx.x == 0) {
        float mean = s1[0] / (float)rows;
        stats[c]      = mean;
        stats[DD + c] = s2[0] / (float)rows - mean * mean;
    }
}

__global__ __launch_bounds__(256) void bn_apply_kernel(
    const float* __restrict__ A, const float* __restrict__ Bres,
    const float* __restrict__ sc, const float* __restrict__ bi,
    const float* __restrict__ stats, float* __restrict__ Out,
    _Float16* __restrict__ Out16, int rows)
{
    long i = (long)blockIdx.x * 256 + threadIdx.x;
    if (i >= (long)rows * DD) return;
    int   c = (int)(i & 127);
    float x = A[i] + Bres[i];
    float v = (x - stats[c]) * rsqrtf(stats[DD + c] + 1e-5f) * sc[c] + bi[c];
    Out[i]   = v;
    Out16[i] = (_Float16)v;
}

// ---------------------------------------------------------------------------
// Decoder: one workgroup (256 threads = 8 waves) per batch element.
// embed_a / K / V (3 x 199 x 128 fp32 = 306 KB) live in the WGP's 320 KB LDS
// for all 199 sequential greedy steps. Lk streams (f16) from L2 each step.
// ---------------------------------------------------------------------------
__global__ __launch_bounds__(256) void decode_kernel(
    const float*    __restrict__ Hemb,  // (B*200, 128) encoder embeddings f32
    const _Float16* __restrict__ kvl,   // (B*199, 384) = [K | V | Lk] f16
    const float* __restrict__ pfc,      // (128,128)
    const float* __restrict__ psc,      // (256,128)
    const float* __restrict__ pout,     // (128,128)
    const float* __restrict__ inp,      // (B,200,2) coords
    float* __restrict__ out_cost, float* __restrict__ out_lp,
    float* __restrict__ out_tours)
{
    extern __shared__ float sm[];
    float* sA   = sm;                 // 199*128 : embed_a
    float* sK   = sA + NA * DD;       // 199*128
    float* sV   = sK + NA * DD;       // 199*128
    float* ef   = sV + NA * DD;       // 128 : embed_f
    float* el   = ef + DD;            // 128 : embed_l
    float* g1   = el + DD;            // 128 : graph
    float* qv   = g1 + DD;            // 128 : query
    float* hd   = qv + DD;            // 128 : heads
    float* gl   = hd + DD;            // 128 : glimpse
    float* sS   = gl + DD;            // 8*199 : per-head scores / attn
    float* lg   = sS + NH * NA;       // 200 : logits
    float* hmax = lg + 200;           // 8
    float* hsum = hmax + 8;           // 8
    float* red  = hsum + 8;           // 256
    float* misc = red + 256;          // 8 : [0]=lp_total [1]=cnt
    int*   redi = (int*)(misc + 8);   // 256
    int*   msk  = redi + 256;         // 200
    int*   tour = msk + 200;          // 200

    const int b = blockIdx.x;
    const int t = threadIdx.x;

    for (int i = t; i < DD; i += 256) { ef[i] = Hemb[((long)b * NPTS) * DD + i]; el[i] = ef[i]; }
    for (int idx = t; idx < NA * DD; idx += 256) {
        int  n = idx >> 7, c = idx & 127;
        sA[idx] = Hemb[((long)b * NPTS + 1 + n) * DD + c];
        long kr = ((long)b * NA + n) * (3 * DD);
        sK[idx] = (float)kvl[kr + c];
        sV[idx] = (float)kvl[kr + DD + c];
    }
    for (int i = t; i < NA; i += 256) msk[i] = 1;
    if (t == 0) { tour[0] = 0; misc[0] = 0.0f; misc[1] = (float)NA; }
    __syncthreads();

    const int w = t >> 5, l = t & 31;

    for (int step = 0; step < NA; ++step) {
        float cnt = misc[1];
        // graph = (sum_masked(embed_a) + embed_f + embed_l) / (cnt + 2)
        if (t < DD) {
            float acc = ef[t] + el[t];
            for (int n = 0; n < NA; ++n)
                if (msk[n]) acc += sA[(n << 7) + t];
            g1[t] = acc / (cnt + 2.0f);
        }
        __syncthreads();
        // q = graph @ pfc + [embed_f, embed_l] @ psc
        if (t < DD) {
            float acc = 0.0f;
            for (int e = 0; e < DD; ++e) acc += g1[e] * pfc[(e << 7) + t];
            for (int e = 0; e < DD; ++e) acc += ef[e] * psc[(e << 7) + t];
            for (int e = 0; e < DD; ++e) acc += el[e] * psc[((e + DD) << 7) + t];
            qv[t] = acc;
        }
        __syncthreads();
        // masked scores per head
        for (int idx = t; idx < NH * NA; idx += 256) {
            int hh = idx / NA, n = idx % NA;
            const float* qp = qv + hh * DKH;
            const float* kp = sK + (n << 7) + hh * DKH;
            float acc = 0.0f;
#pragma unroll
            for (int j = 0; j < DKH; ++j) acc += qp[j] * kp[j];
            sS[idx] = msk[n] ? acc * 0.25f : NEGV;
        }
        __syncthreads();
        // per-head softmax stats (wave w handles head w)
        {
            float m = -3.4e38f;
            for (int n = l; n < NA; n += 32) m = fmaxf(m, sS[w * NA + n]);
            for (int off = 16; off > 0; off >>= 1) m = fmaxf(m, __shfl_xor(m, off, 32));
            float s = 0.0f;
            for (int n = l; n < NA; n += 32) s += __expf(sS[w * NA + n] - m);
            for (int off = 16; off > 0; off >>= 1) s += __shfl_xor(s, off, 32);
            if (l == 0) { hmax[w] = m; hsum[w] = s; }
        }
        __syncthreads();
        // normalize into attn weights
        for (int idx = t; idx < NH * NA; idx += 256) {
            int hh = idx / NA;
            sS[idx] = __expf(sS[idx] - hmax[hh]) / hsum[hh];
        }
        __syncthreads();
        // heads = attn @ V
        if (t < DD) {
            int hh = t >> 4, j = t & 15;
            float acc = 0.0f;
            for (int n = 0; n < NA; ++n) acc += sS[hh * NA + n] * sV[(n << 7) + hh * DKH + j];
            hd[t] = acc;
        }
        __syncthreads();
        // glimpse = heads @ pout
        if (t < DD) {
            float acc = 0.0f;
            for (int e = 0; e < DD; ++e) acc += hd[e] * pout[(e << 7) + t];
            gl[t] = acc;
        }
        __syncthreads();
        // logits = mask( tanh(glimpse . Lk / sqrt(128)) * 10 )
        if (t < NA) {
            const _Float16* lkp = kvl + ((long)b * NA + t) * (3 * DD) + 2 * DD;
            float acc = 0.0f;
            for (int d = 0; d < DD; ++d) acc += gl[d] * (float)lkp[d];
            acc = tanhf(acc * 0.08838834764831845f) * 10.0f;
            lg[t] = msk[t] ? acc : NEGV;
        }
        __syncthreads();
        // argmax (first-max tie break) + logsumexp
        {
            float v  = (t < NA) ? lg[t] : -3.4e38f;
            int   id = (t < NA) ? t : 0x7fffffff;
            red[t] = v; redi[t] = id;
            __syncthreads();
            for (int st = 128; st > 0; st >>= 1) {
                if (t < st) {
                    float v2 = red[t + st]; int i2 = redi[t + st];
                    if (v2 > red[t] || (v2 == red[t] && i2 < redi[t])) { red[t] = v2; redi[t] = i2; }
                }
                __syncthreads();
            }
            float M = red[0]; int sel = redi[0];
            __syncthreads();
            red[t] = (t < NA) ? __expf(lg[t] - M) : 0.0f;
            __syncthreads();
            for (int st = 128; st > 0; st >>= 1) {
                if (t < st) red[t] += red[t + st];
                __syncthreads();
            }
            if (t == 0) {
                misc[0] += lg[sel] - M - __logf(red[0]);
                tour[step + 1] = sel + 1;
                msk[sel] = 0;
                misc[1]  = cnt - 1.0f;
            }
            __syncthreads();
            if (t < DD) el[t] = sA[(sel << 7) + t];
            __syncthreads();
        }
    }

    // tour cost
    float partial = 0.0f;
    for (int i = t; i < NPTS; i += 256) {
        int a = tour[i], c2 = tour[(i + 1) % NPTS];
        float dx = inp[((long)b * NPTS + a) * 2]     - inp[((long)b * NPTS + c2) * 2];
        float dy = inp[((long)b * NPTS + a) * 2 + 1] - inp[((long)b * NPTS + c2) * 2 + 1];
        partial += sqrtf(dx * dx + dy * dy);
    }
    red[t] = partial;
    __syncthreads();
    for (int st = 128; st > 0; st >>= 1) {
        if (t < st) red[t] += red[t + st];
        __syncthreads();
    }
    if (t == 0) { out_cost[b] = red[0]; out_lp[b] = misc[0]; }
    for (int i = t; i < NPTS; i += 256) out_tours[(long)b * NPTS + i] = (float)tour[i];
}

// ---------------------------------------------------------------------------
// Host driver
// ---------------------------------------------------------------------------
static void launch_gemm(const _Float16* A, const _Float16* Bt, const float* bias,
                        float* C, _Float16* C16,
                        int M, int N, int K, int relu, int remap, hipStream_t s)
{
    dim3 grid(M / 16, N / 128), block(64);
    gemm_wmma_kernel<<<grid, block, 0, s>>>(A, Bt, bias, C, C16, M, N, K, relu, remap);
}

extern "C" void kernel_launch(void* const* d_in, const int* in_sizes, int n_in,
                              void* d_out, int out_size, void* d_ws, size_t ws_size,
                              hipStream_t stream)
{
    const float* inp   = (const float*)d_in[0];
    // d_in[1] = n_rollout (== 1 per setup; tiling is identity)
    const float* embW  = (const float*)d_in[2];
    const float* embB  = (const float*)d_in[3];
    const float* Wq    = (const float*)d_in[4];
    const float* Wk    = (const float*)d_in[5];
    const float* Wv    = (const float*)d_in[6];
    const float* Wo    = (const float*)d_in[7];
    const float* bn1s  = (const float*)d_in[8];
    const float* bn1b  = (const float*)d_in[9];
    const float* bn2s  = (const float*)d_in[10];
    const float* bn2b  = (const float*)d_in[11];
    const float* ffW1  = (const float*)d_in[12];
    const float* ffb1  = (const float*)d_in[13];
    const float* ffW2  = (const float*)d_in[14];
    const float* ffb2  = (const float*)d_in[15];
    const float* pneW  = (const float*)d_in[16];
    const float* pfcW  = (const float*)d_in[17];
    const float* pscW  = (const float*)d_in[18];
    const float* poutW = (const float*)d_in[19];

    const long SZ = (long)BNROW * DD;  // 13.1M elements
    float* ws    = (float*)d_ws;
    float* h     = ws;                 // f32 embeddings / bn output
    float* bt    = h + SZ;             // f32 residual branch (Wo out / ff2 out)
    float* stats = bt + SZ;            // 256 f32

    _Float16* f16b = (_Float16*)(stats + 256);   // 16B-aligned (offset mult. of 16)
    _Float16* h16  = f16b;                       // f16 mirror of h
    _Float16* bq16 = h16 + SZ;
    _Float16* bk16 = bq16 + SZ;
    _Float16* bv16 = bk16 + SZ;
    _Float16* bo16 = bv16 + SZ;
    _Float16* fm16 = bo16 + SZ;                  // FF mid (CHM*HIDD == SZ)
    _Float16* wt16 = fm16 + SZ;                  // transposed f16 weights
    _Float16* kvl16 = bq16;                      // decode [K|V|Lk] reuses bq..bv

    _Float16* wqT  = wt16;                       // 3 * D*D each
    _Float16* wkT  = wqT + (long)NLAY * DD * DD;
    _Float16* wvT  = wkT + (long)NLAY * DD * DD;
    _Float16* woT  = wvT + (long)NLAY * DD * DD;
    _Float16* w1T  = woT + (long)NLAY * DD * DD;    // 3 * (HIDD x D)
    _Float16* w2T  = w1T + (long)NLAY * DD * HIDD;  // 3 * (D x HIDD)
    _Float16* pneT = w2T + (long)NLAY * HIDD * DD;  // (3D x D)

    auto tr = [&](const float* W, _Float16* Wt, int K, int N) {
        long tot = (long)K * N;
        transpose_f16_kernel<<<dim3((unsigned)((tot + 255) / 256)), dim3(256), 0, stream>>>(
            W, Wt, K, N);
    };

    // --- one-time (per call) weight transposition to f16 N x K ---
    for (int lyr = 0; lyr < NLAY; ++lyr) {
        tr(Wq   + (long)lyr * DD * DD,   wqT + (long)lyr * DD * DD,   DD, DD);
        tr(Wk   + (long)lyr * DD * DD,   wkT + (long)lyr * DD * DD,   DD, DD);
        tr(Wv   + (long)lyr * DD * DD,   wvT + (long)lyr * DD * DD,   DD, DD);
        tr(Wo   + (long)lyr * DD * DD,   woT + (long)lyr * DD * DD,   DD, DD);
        tr(ffW1 + (long)lyr * DD * HIDD, w1T + (long)lyr * DD * HIDD, DD, HIDD);
        tr(ffW2 + (long)lyr * HIDD * DD, w2T + (long)lyr * HIDD * DD, HIDD, DD);
    }
    tr(pneW, pneT, DD, 3 * DD);

    // --- embedding ---
    {
        long total = (long)BNROW * DD;
        embed_kernel<<<dim3((unsigned)((total + 255) / 256)), dim3(256), 0, stream>>>(
            inp, embW, embB, h, h16, BNROW);
    }

    // --- encoder layers ---
    for (int lyr = 0; lyr < NLAY; ++lyr) {
        launch_gemm(h16, wqT + (long)lyr * DD * DD, nullptr, nullptr, bq16,
                    BNROW, DD, DD, 0, 0, stream);
        launch_gemm(h16, wkT + (long)lyr * DD * DD, nullptr, nullptr, bk16,
                    BNROW, DD, DD, 0, 0, stream);
        launch_gemm(h16, wvT + (long)lyr * DD * DD, nullptr, nullptr, bv16,
                    BNROW, DD, DD, 0, 0, stream);

        attn_kernel<<<dim3(BB * NH), dim3(256), 0, stream>>>(bq16, bk16, bv16, bo16);

        launch_gemm(bo16, woT + (long)lyr * DD * DD, nullptr, bt, nullptr,
                    BNROW, DD, DD, 0, 0, stream);

        bn_reduce_kernel<<<dim3(DD), dim3(256), 0, stream>>>(h, bt, stats, BNROW);
        {
            long total = (long)BNROW * DD;
            bn_apply_kernel<<<dim3((unsigned)((total + 255) / 256)), dim3(256), 0, stream>>>(
                h, bt, bn1s + lyr * DD, bn1b + lyr * DD, stats, h, h16, BNROW);
        }

        // FF in 4 M-chunks; mid activation kept f16-only
        for (int c = 0; c < 4; ++c) {
            launch_gemm(h16 + (long)c * CHM * DD, w1T + (long)lyr * DD * HIDD,
                        ffb1 + lyr * HIDD, nullptr, fm16,
                        CHM, HIDD, DD, /*relu=*/1, 0, stream);
            launch_gemm(fm16, w2T + (long)lyr * HIDD * DD, ffb2 + lyr * DD,
                        bt + (long)c * CHM * DD, nullptr,
                        CHM, DD, HIDD, 0, 0, stream);
        }

        bn_reduce_kernel<<<dim3(DD), dim3(256), 0, stream>>>(h, bt, stats, BNROW);
        {
            long total = (long)BNROW * DD;
            bn_apply_kernel<<<dim3((unsigned)((total + 255) / 256)), dim3(256), 0, stream>>>(
                h, bt, bn2s + lyr * DD, bn2b + lyr * DD, stats, h, h16, BNROW);
        }
    }

    // --- decoder precompute: kvl16 = embed_a @ pne_W  (row remap skips node 0) ---
    launch_gemm(h16, pneT, nullptr, nullptr, kvl16, BB * NA, 3 * DD, DD, 0, /*remap=*/1, stream);

    // --- decode (persistent block per batch element, 199 greedy steps) ---
    float* outF = (float*)d_out;
    size_t smem = (size_t)(3 * NA * DD + 6 * DD + NH * NA + 200 + 16 + 256 + 8) * 4
                + (size_t)(256 + 200 + 200) * 4;  // ~319.6 KB < 320 KB WGP LDS
    (void)hipFuncSetAttribute((const void*)decode_kernel,
                              hipFuncAttributeMaxDynamicSharedMemorySize, (int)smem);
    decode_kernel<<<dim3(BB), dim3(256), smem, stream>>>(
        h, kvl16, pfcW, pscW, poutW, inp,
        outF, outF + BB, outF + 2 * BB);

    (void)in_sizes; (void)n_in; (void)out_size; (void)ws_size;
}
